// GraphEncoder_83330955477963
// MI455X (gfx1250) — matrix operand order repaired and checked by reading, not compiled, not run on hardware
//
#include <hip/hip_runtime.h>
#include <hip/hip_bf16.h>
#include <math.h>

// Problem constants (match reference)
#define DDIM  128
#define HHEADS 8
#define CCH   16
#define FRAW  4
#define NEG_SLOPE 0.2f
#define BN_EPS 1e-5f
#define SM_EPS 1e-16f

#define MROWS 64            // output rows per block in the WMMA GEMM
#define LDA_PAD 136         // f16 row stride (272B = 17*16B: keeps 16B align, dodges conflicts)

typedef __attribute__((ext_vector_type(16))) _Float16 v16h;
typedef __attribute__((ext_vector_type(8)))  _Float16 v8h;
typedef __attribute__((ext_vector_type(8)))  float    v8f;

// ---------------------------------------------------------------------------
// h0 = x @ W0   (N x 4) @ (4 x 128): tiny K, per-thread dot. Also writes the
// f16 shadow of h used by the WMMA transform.
// ---------------------------------------------------------------------------
__global__ void k_input_gemm(const float* __restrict__ x,
                             const float* __restrict__ W0,
                             float* __restrict__ h,
                             _Float16* __restrict__ h16, int n) {
    int idx = blockIdx.x * blockDim.x + threadIdx.x;
    if (idx >= n * DDIM) return;
    int node = idx / DDIM, d = idx % DDIM;
    float acc = 0.f;
#pragma unroll
    for (int f = 0; f < FRAW; ++f) acc += x[node * FRAW + f] * W0[f * DDIM + d];
    h[idx]   = acc;
    h16[idx] = (_Float16)acc;
}

// ---------------------------------------------------------------------------
// Wt16[n][k] = (f16) W[k][n]  -- transposed f16 weights, once per layer.
// ---------------------------------------------------------------------------
__global__ void k_convert_w(const float* __restrict__ W, _Float16* __restrict__ Wt16) {
    int idx = blockIdx.x * blockDim.x + threadIdx.x;
    if (idx >= DDIM * DDIM) return;
    int nn = idx >> 7, k = idx & 127;
    Wt16[idx] = (_Float16)W[k * DDIM + nn];
}

// ---------------------------------------------------------------------------
// hw = h @ W   via f16 WMMA, f32 accumulate.
// Block = 256 threads = 8 waves. Block computes 64 rows x 128 cols.
// Each wave owns a 16-col stripe and 4 M-tiles (16 WMMAs), K=128 in 4 chunks.
// A rows and transposed W staged in LDS with gfx1250 async-to-LDS copies.
// ---------------------------------------------------------------------------
__global__ __launch_bounds__(256)
void k_gemm_wmma(const _Float16* __restrict__ h16,
                 const _Float16* __restrict__ wt16,
                 float* __restrict__ hw, int n) {
    __shared__ _Float16 ldsA[MROWS][LDA_PAD];   // [m][k]
    __shared__ _Float16 ldsW[DDIM][LDA_PAD];    // [n][k] (W transposed)

    const int tid  = threadIdx.x;
    const int row0 = blockIdx.x * MROWS;
    const int rr   = tid >> 4;    // 0..15: row within 16-row staging group
    const int seg  = tid & 15;    // 16B segment within a 256B row

    // ---- async stage W^T: 128 rows x 256B (8 iterations x 256 lanes x 16B)
#pragma unroll
    for (int it = 0; it < 8; ++it) {
        int r = it * 16 + rr;
        const _Float16* gsrc = wt16 + r * DDIM + seg * 8;
        unsigned ldst = (unsigned)(uintptr_t)&ldsW[r][seg * 8];  // low 32b = LDS offset
        asm volatile("global_load_async_to_lds_b128 %0, %1, off"
                     :: "v"(ldst), "v"(gsrc) : "memory");
    }
    // ---- async stage A: 64 rows x 256B (h16 is padded to a multiple of 64 rows)
#pragma unroll
    for (int it = 0; it < 4; ++it) {
        int r = it * 16 + rr;
        const _Float16* gsrc = h16 + (size_t)(row0 + r) * DDIM + seg * 8;
        unsigned ldst = (unsigned)(uintptr_t)&ldsA[r][seg * 8];
        asm volatile("global_load_async_to_lds_b128 %0, %1, off"
                     :: "v"(ldst), "v"(gsrc) : "memory");
    }
    asm volatile("s_wait_asynccnt 0" ::: "memory");
    __syncthreads();

    const int lane = tid & 31;    // wave32
    const int wave = tid >> 5;
    const int n0   = wave * 16;   // this wave's output column tile
    const int g    = lane >> 4;   // lane group (0/1)
    const int ml   = lane & 15;   // M within tile for A, N for B/C

    v8f acc[4] = {};
#pragma unroll
    for (int kt = 0; kt < 4; ++kt) {
        const int kb = kt * 32;
        // B 32x16 f16 fragment: lane group g holds K kb+g*16 .. kb+g*16+15,
        // contiguous in ldsW[n][k] -> two 16B ds loads.
        v8h blo = *(const v8h*)&ldsW[n0 + ml][kb + g * 16];
        v8h bhi = *(const v8h*)&ldsW[n0 + ml][kb + g * 16 + 8];
        v16h b  = __builtin_shufflevector(blo, bhi,
                    0,1,2,3,4,5,6,7,8,9,10,11,12,13,14,15);
#pragma unroll
        for (int mt = 0; mt < 4; ++mt) {
            const int mrow = mt * 16 + ml;
            // A 16x32 f16 fragment: lane group g holds K kb+g*16 .. kb+g*16+15 (matches B)
            v8h alo = *(const v8h*)&ldsA[mrow][kb + g * 16];
            v8h ahi = *(const v8h*)&ldsA[mrow][kb + g * 16 + 8];
            v16h a  = __builtin_shufflevector(alo, ahi,
                        0,1,2,3,4,5,6,7,8,9,10,11,12,13,14,15);
            acc[mt] = __builtin_amdgcn_wmma_f32_16x16x32_f16(
                false, a, false, b, (short)0, acc[mt], false, false);
        }
    }

    // C/D layout: VGPR r -> M = r + 8*(lane/16), N = lane%16
#pragma unroll
    for (int mt = 0; mt < 4; ++mt) {
#pragma unroll
        for (int r = 0; r < 8; ++r) {
            int gr = row0 + mt * 16 + r + 8 * g;
            if (gr < n) hw[(size_t)gr * DDIM + n0 + ml] = acc[mt][r];
        }
    }
}

// ---------------------------------------------------------------------------
// Per-node, per-head attention logit contributions
// ---------------------------------------------------------------------------
__global__ void k_logits(const float* __restrict__ hw,
                         const float* __restrict__ a_src,
                         const float* __restrict__ a_dst,
                         float* __restrict__ es, float* __restrict__ ed, int n) {
    int idx = blockIdx.x * blockDim.x + threadIdx.x;
    if (idx >= n * HHEADS) return;
    int node = idx / HHEADS, hd = idx % HHEADS;
    const float* p = hw + (size_t)node * DDIM + hd * CCH;
    float s = 0.f, d = 0.f;
#pragma unroll
    for (int c = 0; c < CCH; ++c) {
        float v = p[c];
        s += v * a_src[hd * CCH + c];
        d += v * a_dst[hd * CCH + c];
    }
    es[idx] = s;
    ed[idx] = d;
}

// ---------------------------------------------------------------------------
// Zero-init scratch
// ---------------------------------------------------------------------------
__global__ void k_fill(float* __restrict__ acc, unsigned* __restrict__ nmax,
                       float* __restrict__ nsum, float* __restrict__ bns, int n) {
    int idx = blockIdx.x * blockDim.x + threadIdx.x;
    if (idx < n * DDIM) acc[idx] = 0.f;
    if (idx < n * HHEADS) { nmax[idx] = 0u; nsum[idx] = 0.f; }
    if (idx < 2 * DDIM) bns[idx] = 0.f;
}

// Order-preserving float<->uint map so atomicMax(uint) implements float max.
__device__ __forceinline__ unsigned f2ord(float f) {
    unsigned u = __float_as_uint(f);
    return (u & 0x80000000u) ? ~u : (u | 0x80000000u);
}
__device__ __forceinline__ float ord2f(unsigned u) {
    return (u & 0x80000000u) ? __uint_as_float(u & 0x7FFFFFFFu)
                             : __uint_as_float(~u);
}

__device__ __forceinline__ void edge_sd(const int* __restrict__ ei, int E, int e,
                                        int& s, int& d) {
    if (e < E) { s = ei[e]; d = ei[E + e]; }  // edge_index rows: [0]=src [1]=dst
    else       { s = d = e - E; }             // appended self loops
}

// segment_max over incoming edges (thread per edge*head)
__global__ void k_edge_max(const int* __restrict__ ei, int E, int n,
                           const float* __restrict__ es,
                           const float* __restrict__ ed,
                           unsigned* __restrict__ nmax) {
    int idx = blockIdx.x * blockDim.x + threadIdx.x;
    int tot = (E + n) * HHEADS;
    if (idx >= tot) return;
    int e = idx / HHEADS, hd = idx % HHEADS;
    int s, d; edge_sd(ei, E, e, s, d);
    float v = es[s * HHEADS + hd] + ed[d * HHEADS + hd];
    v = v > 0.f ? v : NEG_SLOPE * v;
    atomicMax(&nmax[d * HHEADS + hd], f2ord(v));
}

// exp(e - max[dst]) with per-edge stash + segment_sum
__global__ void k_edge_exp(const int* __restrict__ ei, int E, int n,
                           const float* __restrict__ es,
                           const float* __restrict__ ed,
                           const unsigned* __restrict__ nmax,
                           float* __restrict__ nsum, float* __restrict__ eexp) {
    int idx = blockIdx.x * blockDim.x + threadIdx.x;
    int tot = (E + n) * HHEADS;
    if (idx >= tot) return;
    int e = idx / HHEADS, hd = idx % HHEADS;
    int s, d; edge_sd(ei, E, e, s, d);
    float v = es[s * HHEADS + hd] + ed[d * HHEADS + hd];
    v = v > 0.f ? v : NEG_SLOPE * v;
    float m = ord2f(nmax[d * HHEADS + hd]);
    float ex = __expf(v - m);
    eexp[idx] = ex;
    atomicAdd(&nsum[d * HHEADS + hd], ex);
}

// message aggregation: block per edge, thread per channel (coalesced hw[src])
__global__ __launch_bounds__(DDIM)
void k_aggregate(const int* __restrict__ ei, int E, int n,
                 const float* __restrict__ hw, const float* __restrict__ eexp,
                 const float* __restrict__ nsum, float* __restrict__ acc) {
    int e = blockIdx.x;
    int c = threadIdx.x;
    int s, d; edge_sd(ei, E, e, s, d);
    int hd = c >> 4;
    float alpha = eexp[e * HHEADS + hd] / (nsum[d * HHEADS + hd] + SM_EPS);
    atomicAdd(&acc[(size_t)d * DDIM + c], hw[(size_t)s * DDIM + c] * alpha);
}

// residual + bias, plus BN partial sums (one channel per lane, 256 rows/block)
#define ROWS_PER_BLOCK 256
__global__ __launch_bounds__(DDIM)
void k_residual_stats(const float* __restrict__ h, const float* __restrict__ acc,
                      const float* __restrict__ bias, float* __restrict__ hres,
                      float* __restrict__ bns, int n) {
    int c  = threadIdx.x;
    int r0 = blockIdx.x * ROWS_PER_BLOCK;
    int r1 = r0 + ROWS_PER_BLOCK; if (r1 > n) r1 = n;
    float bc = bias[c];
    float s = 0.f, sq = 0.f;
    for (int r = r0; r < r1; ++r) {
        float v = h[(size_t)r * DDIM + c] + acc[(size_t)r * DDIM + c] + bc;
        hres[(size_t)r * DDIM + c] = v;
        s += v; sq += v * v;
    }
    atomicAdd(&bns[c], s);
    atomicAdd(&bns[DDIM + c], sq);
}

// apply BN (training mode, biased variance); refreshes f16 shadow of h
__global__ void k_bn_apply(const float* __restrict__ hres,
                           const float* __restrict__ bns,
                           const float* __restrict__ gamma,
                           const float* __restrict__ beta,
                           float* __restrict__ out,
                           _Float16* __restrict__ h16, int n) {
    int idx = blockIdx.x * blockDim.x + threadIdx.x;
    if (idx >= n * DDIM) return;
    int c = idx & (DDIM - 1);
    float inv_n = 1.0f / (float)n;
    float mean  = bns[c] * inv_n;
    float var   = bns[DDIM + c] * inv_n - mean * mean;
    float v = gamma[c] * (hres[idx] - mean) * rsqrtf(var + BN_EPS) + beta[c];
    out[idx] = v;
    h16[idx] = (_Float16)v;
}

// ---------------------------------------------------------------------------
// Host orchestration
// ---------------------------------------------------------------------------
static inline size_t align256(size_t x) { return (x + 255) & ~(size_t)255; }

extern "C" void kernel_launch(void* const* d_in, const int* in_sizes, int n_in,
                              void* d_out, int out_size, void* d_ws, size_t ws_size,
                              hipStream_t stream) {
    const float* x    = (const float*)d_in[0];
    const int*   ei   = (const int*)  d_in[1];
    const float* W0   = (const float*)d_in[2];
    const float* Wl   = (const float*)d_in[3];   // [3,128,128]
    const float* asrc = (const float*)d_in[4];   // [3,8,16]
    const float* adst = (const float*)d_in[5];   // [3,8,16]
    const float* bias = (const float*)d_in[6];   // [3,128]
    const float* gam  = (const float*)d_in[7];   // [3,128]
    const float* bet  = (const float*)d_in[8];   // [3,128]

    const int N = in_sizes[0] / FRAW;
    const int E = in_sizes[1] / 2;
    const int TOT_E = E + N;                     // with self loops
    const int NPAD  = (N + MROWS - 1) & ~(MROWS - 1);  // async staging never OOB

    // workspace layout
    char* p = (char*)d_ws;
    float* bufA = (float*)p; p += align256((size_t)N * DDIM * 4);      // h (f32)
    float* bufB = (float*)p; p += align256((size_t)N * DDIM * 4);      // hw / hres
    float* bufC = (float*)p; p += align256((size_t)N * DDIM * 4);      // GAT accum
    _Float16* h16  = (_Float16*)p; p += align256((size_t)NPAD * DDIM * 2); // f16 shadow of h
    _Float16* Wt16 = (_Float16*)p; p += align256((size_t)DDIM * DDIM * 2); // f16 W^T
    float* es   = (float*)p; p += align256((size_t)N * HHEADS * 4);
    float* ed   = (float*)p; p += align256((size_t)N * HHEADS * 4);
    unsigned* nmax = (unsigned*)p; p += align256((size_t)N * HHEADS * 4);
    float* nsum = (float*)p; p += align256((size_t)N * HHEADS * 4);
    float* eexp = (float*)p; p += align256((size_t)TOT_E * HHEADS * 4);
    float* bns  = (float*)p; p += align256((size_t)2 * DDIM * 4);
    (void)ws_size; (void)n_in; (void)out_size;

    const int TB = 256;
    const int g_nd   = (N * DDIM + TB - 1) / TB;
    const int g_nh   = (N * HHEADS + TB - 1) / TB;
    const int g_eh   = (TOT_E * HHEADS + TB - 1) / TB;
    const int g_rows = (N + ROWS_PER_BLOCK - 1) / ROWS_PER_BLOCK;
    const int g_m64  = (N + MROWS - 1) / MROWS;
    const int g_w    = (DDIM * DDIM + TB - 1) / TB;

    // h0 = x @ W0 (f32 + f16 shadow)
    k_input_gemm<<<g_nd, TB, 0, stream>>>(x, W0, bufA, h16, N);

    for (int l = 0; l < 3; ++l) {
        const float* W  = Wl   + (size_t)l * DDIM * DDIM;
        const float* as = asrc + (size_t)l * HHEADS * CCH;
        const float* ad = adst + (size_t)l * HHEADS * CCH;
        const float* b  = bias + (size_t)l * DDIM;
        const float* g  = gam  + (size_t)l * DDIM;
        const float* be = bet  + (size_t)l * DDIM;

        // W -> f16 transposed (32KB, L2-resident)
        k_convert_w<<<g_w, TB, 0, stream>>>(W, Wt16);
        // hw = h @ W  (WMMA, async-to-LDS staging)
        k_gemm_wmma<<<g_m64, TB, 0, stream>>>(h16, Wt16, bufB, N);
        // attention logit contributions
        k_logits<<<g_nh, TB, 0, stream>>>(bufB, as, ad, es, ed, N);
        // clear scratch
        k_fill<<<g_nd, TB, 0, stream>>>(bufC, nmax, nsum, bns, N);
        // softmax over incoming edges
        k_edge_max<<<g_eh, TB, 0, stream>>>(ei, E, N, es, ed, nmax);
        k_edge_exp<<<g_eh, TB, 0, stream>>>(ei, E, N, es, ed, nmax, nsum, eexp);
        // weighted message aggregation
        k_aggregate<<<TOT_E, DDIM, 0, stream>>>(ei, E, N, bufB, eexp, nsum, bufC);
        // residual + bias + BN stats (hres -> bufB, hw no longer needed)
        k_residual_stats<<<g_rows, DDIM, 0, stream>>>(bufA, bufC, b, bufB, bns, N);
        // BN apply; last layer writes straight to d_out; h16 refreshed for next layer
        float* dst = (l == 2) ? (float*)d_out : bufA;
        k_bn_apply<<<g_nd, TB, 0, stream>>>(bufB, bns, g, be, dst, h16, N);
    }
}
